// EncoderLayer_36979668418684
// MI455X (gfx1250) — compile-verified, matrix operand order checked by reading, not run on hardware
//
#include <hip/hip_runtime.h>
#include <hip/hip_bf16.h>
#include <math.h>

typedef __bf16 bf16_t;
typedef __attribute__((ext_vector_type(16))) __bf16 v16bf;
typedef __attribute__((ext_vector_type(8)))  float  v8f;
typedef unsigned int u32x4 __attribute__((ext_vector_type(4)));
typedef int          i32x8 __attribute__((ext_vector_type(8)));
typedef int          i32x4 __attribute__((ext_vector_type(4)));

constexpr int B_ = 8, S_ = 1024, D_ = 768, H_ = 12, HD_ = 64, F_ = 3072;
constexpr int BS_ = B_ * S_;         // 8192
constexpr int D3_ = 3 * D_;          // 2304

// Padded LDS row pitch for a 32-element (64 B) tile row: +16 B pad -> 80 B.
// Bank check: row r starts at dword r*20; r*20 mod 64 distinct for r=0..15.
constexpr int LDS_ROW = 40;          // elements (80 bytes)

#if defined(__HIP_DEVICE_COMPILE__) && \
    __has_builtin(__builtin_amdgcn_tensor_load_to_lds) && \
    __has_builtin(__builtin_amdgcn_s_wait_tensorcnt)
#define USE_TDM 1
#else
#define USE_TDM 0
#endif

// ---------------------------------------------------------------------------
// Fragment load, CDNA5 16-bit A-operand layout (16x32):
//   lane L: row M = L%16, half = L/16
//   elements 0..7  hold K = half*8 + 0..7
//   elements 8..15 hold K = 16 + half*8 + 0..7
// Identical pattern serves the B operand when B is stored as [N,K] rows.
// Two 16-byte loads -> global_load_b128 / ds_load_b128.
// ---------------------------------------------------------------------------
__device__ inline v16bf load_frag(const bf16_t* __restrict__ base, int row,
                                  int ld, int k0, int lane) {
  const bf16_t* p = base + (size_t)row * (size_t)ld + k0 + ((lane >> 4) << 3);
  union { v16bf v; uint4 u[2]; } r;
  r.u[0] = *reinterpret_cast<const uint4*>(p);
  r.u[1] = *reinterpret_cast<const uint4*>(p + 16);
  return r.v;
}

__device__ inline v8f wmma_bf16(v16bf a, v16bf b, v8f c) {
  return __builtin_amdgcn_wmma_f32_16x16x32_bf16(
      false, a, false, b, (short)0, c, false, false);
}

#if USE_TDM
// ---------------------------------------------------------------------------
// TDM: DMA a 2D tile (tile_k x tile_rows bf16, row stride = stride_elems)
// from global memory into LDS at byte offset lds_off, with 16 B of LDS pad
// inserted after every 64 B row (pad_interval=16 DW, pad_amount=4 DW) so the
// staged tile has an 80 B pitch. Descriptor layout per CDNA5 ISA ch.8.
// ---------------------------------------------------------------------------
__device__ inline void tdm_load_2d(unsigned int lds_off,
                                   const bf16_t* gptr,
                                   int tile_k, int tile_rows,
                                   long stride_elems) {
  unsigned long long ga = (unsigned long long)(uintptr_t)gptr;
  u32x4 g0;
  g0[0] = 1u;                                   // count=1, user load descriptor
  g0[1] = lds_off;                              // lds_addr (bytes)
  g0[2] = (unsigned int)(ga & 0xFFFFFFFFu);     // global_addr[31:0]
  g0[3] = (unsigned int)((ga >> 32) & 0x01FFFFFFu) | (2u << 30);  // type=2
  i32x8 g1;
  g1[0] = (1 << 16)     // data_size = 2 bytes
        | (1 << 20)     // pad_enable
        | (3 << 22)     // pad_interval: 16 dwords
        | (3 << 25);    // pad_amount: 4 dwords (16 B)
  g1[1] = (int)((unsigned int)tile_k << 16);                    // tensor_dim0 lo
  g1[2] = (int)(((unsigned int)tile_k >> 16) |
                ((unsigned int)tile_rows << 16));               // td0 hi | td1 lo
  g1[3] = (int)(((unsigned int)tile_rows >> 16) |
                ((unsigned int)tile_k << 16));                  // td1 hi | tile_dim0
  g1[4] = tile_rows;                                            // tile_dim1 (dim2=0)
  g1[5] = (int)(unsigned int)(stride_elems & 0xFFFFFFFF);       // dim0 stride lo
  g1[6] = (int)(unsigned int)((stride_elems >> 32) & 0xFFFF);   // dim0 stride hi
  g1[7] = 0;
  i32x4 gz = {0, 0, 0, 0};
#if __clang_major__ >= 23
  i32x8 gz8 = {0, 0, 0, 0, 0, 0, 0, 0};
  __builtin_amdgcn_tensor_load_to_lds(g0, g1, gz, gz, gz8, 0);
#else
  __builtin_amdgcn_tensor_load_to_lds(g0, g1, gz, gz, 0);
#endif
}
#endif  // USE_TDM

// ---------------------------------------------------------------------------
// Elementwise f32 -> bf16 cast
// ---------------------------------------------------------------------------
__global__ void cast_f32_bf16_kernel(const float* __restrict__ in,
                                     bf16_t* __restrict__ out, int n) {
  int i = blockIdx.x * blockDim.x + threadIdx.x;
  if (i < n) out[i] = (bf16_t)in[i];
}

// Build Wqkv [2304, 768] ([N,K] rows) from q_w/k_w/v_w [H, D, HD].
__global__ void build_wqkv_kernel(const float* __restrict__ qw,
                                  const float* __restrict__ kw,
                                  const float* __restrict__ vw,
                                  bf16_t* __restrict__ out) {
  int i = blockIdx.x * blockDim.x + threadIdx.x;
  if (i >= D3_ * D_) return;
  int d   = i % D_;
  int n   = i / D_;
  int sec = n / D_;
  int hn  = n % D_;
  int h   = hn / HD_;
  int e   = hn % HD_;
  const float* src = (sec == 0) ? qw : (sec == 1) ? kw : vw;
  out[i] = (bf16_t)src[((size_t)h * D_ + d) * HD_ + e];
}

// vt[b,h,e,t] = qkv[b,t, 2D + h*HD + e]
__global__ void transpose_v_kernel(const bf16_t* __restrict__ qkv,
                                   bf16_t* __restrict__ vt) {
  int i = blockIdx.x * blockDim.x + threadIdx.x;
  if (i >= B_ * H_ * HD_ * S_) return;
  int t = i % S_;
  int e = (i / S_) % HD_;
  int h = (i / (S_ * HD_)) % H_;
  int b =  i / (S_ * HD_ * H_);
  vt[i] = qkv[((size_t)(b * S_ + t)) * D3_ + 2 * D_ + h * HD_ + e];
}

// ---------------------------------------------------------------------------
// bf16 WMMA GEMM with TDM-staged, double-buffered LDS tiles.
//   C[M,N] = A[M,K] * W^T, W stored [N,K] row-major.
// Block = 128 threads = 4 waves; block tile 128x128; wave tile 64x64.
// Per k-step: wave0 issues 2 TDM tile DMAs for the next step, all waves run
// 16 v_wmma from the current LDS buffers, then s_wait_tensorcnt + barrier.
// MODE 0: bf16. MODE 1: +bias, exact GELU, bf16. MODE 2: +bias, f32.
// ---------------------------------------------------------------------------
template <int MODE>
__global__ __launch_bounds__(128)
void gemm_bf16_kernel(const bf16_t* __restrict__ A,
                      const bf16_t* __restrict__ W,
                      const float* __restrict__ bias,
                      void* __restrict__ Cout, int M, int N, int K) {
  __shared__ bf16_t sA[2][128 * LDS_ROW];   // 2 x 10240 B
  __shared__ bf16_t sW[2][128 * LDS_ROW];   // 2 x 10240 B

  const int lane  = threadIdx.x & 31;
  const int wave  = threadIdx.x >> 5;
  const int l15   = lane & 15;
  const int half8 = (lane >> 4) << 3;
  const int mb = blockIdx.y * 128;
  const int nb = blockIdx.x * 128;
  const int wm = (wave >> 1) * 64;          // wave offset inside block tile
  const int wn = (wave & 1) * 64;

  v8f acc[4][4];
  v8f zero = {};
#pragma unroll
  for (int i = 0; i < 4; ++i)
#pragma unroll
    for (int j = 0; j < 4; ++j) acc[i][j] = zero;

  const int nk = K >> 5;

  auto stage = [&](int kt, int b) {
#if USE_TDM
    if (wave == 0) {
      tdm_load_2d((unsigned int)(uintptr_t)&sA[b][0],
                  A + (size_t)mb * K + kt * 32, 32, 128, K);
      tdm_load_2d((unsigned int)(uintptr_t)&sW[b][0],
                  W + (size_t)nb * K + kt * 32, 32, 128, K);
    }
#else
    // cooperative fallback: each thread copies one padded row of A and W
    int r = threadIdx.x;
    const bf16_t* ga = A + (size_t)(mb + r) * K + kt * 32;
    const bf16_t* gw = W + (size_t)(nb + r) * K + kt * 32;
    bf16_t* la = &sA[b][r * LDS_ROW];
    bf16_t* lw = &sW[b][r * LDS_ROW];
#pragma unroll
    for (int c = 0; c < 4; ++c) {
      *reinterpret_cast<uint4*>(la + c * 8) =
          *reinterpret_cast<const uint4*>(ga + c * 8);
      *reinterpret_cast<uint4*>(lw + c * 8) =
          *reinterpret_cast<const uint4*>(gw + c * 8);
    }
#endif
  };
  auto wait_stage = [&]() {
#if USE_TDM
    if (wave == 0) __builtin_amdgcn_s_wait_tensorcnt(0);
#endif
    __syncthreads();
  };

  stage(0, 0);
  wait_stage();

  for (int kt = 0; kt < nk; ++kt) {
    const int cur = kt & 1;
    if (kt + 1 < nk) stage(kt + 1, cur ^ 1);   // overlap DMA with WMMA

    v16bf af[4], wf[4];
#pragma unroll
    for (int i = 0; i < 4; ++i)
      af[i] = load_frag(&sA[cur][0], wm + 16 * i + l15, LDS_ROW, 0, lane);
#pragma unroll
    for (int j = 0; j < 4; ++j)
      wf[j] = load_frag(&sW[cur][0], wn + 16 * j + l15, LDS_ROW, 0, lane);
#pragma unroll
    for (int i = 0; i < 4; ++i)
#pragma unroll
      for (int j = 0; j < 4; ++j) acc[i][j] = wmma_bf16(af[i], wf[j], acc[i][j]);

    wait_stage();
  }

#pragma unroll
  for (int i = 0; i < 4; ++i) {
#pragma unroll
    for (int j = 0; j < 4; ++j) {
      int col = nb + wn + 16 * j + l15;
      float bv = (MODE == 0) ? 0.0f : bias[col];
#pragma unroll
      for (int r = 0; r < 8; ++r) {
        int row = mb + wm + 16 * i + r + half8;
        float v = acc[i][j][r];
        if (MODE == 0) {
          ((bf16_t*)Cout)[(size_t)row * N + col] = (bf16_t)v;
        } else if (MODE == 1) {
          v += bv;
          v = 0.5f * v * (1.0f + erff(v * 0.70710678118654752f));  // exact GELU
          ((bf16_t*)Cout)[(size_t)row * N + col] = (bf16_t)v;
        } else {
          ((float*)Cout)[(size_t)row * N + col] = v + bv;
        }
      }
    }
  }
}

// ---------------------------------------------------------------------------
// Flash-style attention. Grid: (S/64, B*H), block 128 = 4 waves.
// Wave owns 16 query rows: QK^T (2 WMMA / 16 keys), online softmax with
// width-16 shuffle reductions, P->LDS->A-frag reshape, PV (4 WMMA / 32 keys).
// ---------------------------------------------------------------------------
__global__ __launch_bounds__(128)
void attn_kernel(const bf16_t* __restrict__ qkv, const bf16_t* __restrict__ vt,
                 const unsigned char* __restrict__ mask,
                 bf16_t* __restrict__ wv) {
  __shared__ bf16_t Ptile[4][16 * 32];

  const int lane = threadIdx.x & 31;
  const int wave = threadIdx.x >> 5;
  const int half = lane >> 4;
  const int l15  = lane & 15;
  const int bh = blockIdx.y;
  const int b  = bh / H_;
  const int h  = bh % H_;
  const int q0 = blockIdx.x * 64 + wave * 16;

  const bf16_t* qbase = qkv + (size_t)b * S_ * D3_ + h * HD_;
  const bf16_t* kbase = qbase + D_;
  const bf16_t* vbase = vt + (size_t)bh * HD_ * S_;
  const unsigned char* mbase = mask + (size_t)b * S_ * S_;

  v16bf aq0 = load_frag(qbase, q0 + l15, D3_, 0, lane);
  v16bf aq1 = load_frag(qbase, q0 + l15, D3_, 32, lane);

  v8f accf[4];
  v8f zero = {};
#pragma unroll
  for (int nf = 0; nf < 4; ++nf) accf[nf] = zero;
  float mrow[8], lrow[8];
#pragma unroll
  for (int r = 0; r < 8; ++r) { mrow[r] = -3.0e38f; lrow[r] = 0.0f; }

  for (int pair = 0; pair < S_ / 32; ++pair) {
    v8f sf[2];
#pragma unroll
    for (int sub = 0; sub < 2; ++sub) {
      int t0 = pair * 32 + sub * 16;
      v16bf kf0 = load_frag(kbase, t0 + l15, D3_, 0, lane);
      v16bf kf1 = load_frag(kbase, t0 + l15, D3_, 32, lane);
      v8f c = zero;
      c = wmma_bf16(aq0, kf0, c);
      c = wmma_bf16(aq1, kf1, c);
      int tcol = t0 + l15;
#pragma unroll
      for (int r = 0; r < 8; ++r) {
        float sv = c[r] * 0.125f;                 // 1/sqrt(64)
        int srow = q0 + r + half * 8;
        if (!mbase[(size_t)srow * S_ + tcol]) sv = -1.0e9f;
        c[r] = sv;
      }
      sf[sub] = c;
    }

    float fcorr[8];
#pragma unroll
    for (int r = 0; r < 8; ++r) {
      float t = fmaxf(sf[0][r], sf[1][r]);
#pragma unroll
      for (int off = 8; off >= 1; off >>= 1) t = fmaxf(t, __shfl_xor(t, off, 16));
      float mo = mrow[r];
      float mn = fmaxf(mo, t);
      float f  = __expf(mo - mn);
      float p0 = __expf(sf[0][r] - mn);
      float p1 = __expf(sf[1][r] - mn);
      sf[0][r] = p0;
      sf[1][r] = p1;
      float s = p0 + p1;
#pragma unroll
      for (int off = 8; off >= 1; off >>= 1) s += __shfl_xor(s, off, 16);
      lrow[r] = lrow[r] * f + s;
      mrow[r] = mn;
      fcorr[r] = f;
    }
#pragma unroll
    for (int nf = 0; nf < 4; ++nf)
#pragma unroll
      for (int r = 0; r < 8; ++r) accf[nf][r] *= fcorr[r];

    bf16_t* P = Ptile[wave];
#pragma unroll
    for (int sub = 0; sub < 2; ++sub)
#pragma unroll
      for (int r = 0; r < 8; ++r) {
        int prow = r + half * 8;
        P[prow * 32 + sub * 16 + l15] = (bf16_t)sf[sub][r];
      }
    __syncthreads();

    v16bf pf;
    {
      const bf16_t* pp = P + l15 * 32 + half * 8;
      union { v16bf v; uint4 u[2]; } rr;
      rr.u[0] = *reinterpret_cast<const uint4*>(pp);
      rr.u[1] = *reinterpret_cast<const uint4*>(pp + 16);
      pf = rr.v;
    }
#pragma unroll
    for (int nf = 0; nf < 4; ++nf) {
      v16bf vf = load_frag(vbase, nf * 16 + l15, S_, pair * 32, lane);
      accf[nf] = wmma_bf16(pf, vf, accf[nf]);
    }
    __syncthreads();
  }

#pragma unroll
  for (int nf = 0; nf < 4; ++nf)
#pragma unroll
    for (int r = 0; r < 8; ++r) {
      int srow = q0 + r + half * 8;
      float o = accf[nf][r] / lrow[r];
      wv[((size_t)b * S_ + srow) * D_ + h * HD_ + nf * 16 + l15] = (bf16_t)o;
    }
}

// ---------------------------------------------------------------------------
extern "C" void kernel_launch(void* const* d_in, const int* in_sizes, int n_in,
                              void* d_out, int out_size, void* d_ws,
                              size_t ws_size, hipStream_t stream) {
  const float* x   = (const float*)d_in[0];
  const unsigned char* mask = (const unsigned char*)d_in[1];
  const float* qw  = (const float*)d_in[2];
  const float* kw  = (const float*)d_in[3];
  const float* vw  = (const float*)d_in[4];
  const float* ow  = (const float*)d_in[5];
  const float* f1w = (const float*)d_in[6];
  const float* f1b = (const float*)d_in[7];
  const float* f2w = (const float*)d_in[8];
  const float* f2b = (const float*)d_in[9];

  char* ws = (char*)d_ws;
  auto alloc = [&](size_t bytes) -> void* {
    void* p = (void*)ws;
    ws += (bytes + 255) & ~(size_t)255;
    return p;
  };

  bf16_t* wqkv  = (bf16_t*)alloc((size_t)D3_ * D_ * 2);
  bf16_t* wo    = (bf16_t*)alloc((size_t)D_ * D_ * 2);
  bf16_t* wf1   = (bf16_t*)alloc((size_t)F_ * D_ * 2);
  bf16_t* wf2   = (bf16_t*)alloc((size_t)D_ * F_ * 2);
  bf16_t* x_bf  = (bf16_t*)alloc((size_t)BS_ * D_ * 2);
  bf16_t* qkvb  = (bf16_t*)alloc((size_t)BS_ * D3_ * 2);
  bf16_t* vtb   = (bf16_t*)alloc((size_t)B_ * H_ * HD_ * S_ * 2);
  bf16_t* wvb   = (bf16_t*)alloc((size_t)BS_ * D_ * 2);
  bf16_t* aob   = (bf16_t*)alloc((size_t)BS_ * D_ * 2);
  bf16_t* h1b   = x_bf;  // alias: BS*D*2 + BS*3D*2 == BS*F*2 exactly

  {
    int n = BS_ * D_;
    cast_f32_bf16_kernel<<<(n + 255) / 256, 256, 0, stream>>>(x, x_bf, n);
  }
  {
    int n = D_ * D_;
    cast_f32_bf16_kernel<<<(n + 255) / 256, 256, 0, stream>>>(ow, wo, n);
  }
  {
    int n = F_ * D_;
    cast_f32_bf16_kernel<<<(n + 255) / 256, 256, 0, stream>>>(f1w, wf1, n);
    cast_f32_bf16_kernel<<<(n + 255) / 256, 256, 0, stream>>>(f2w, wf2, n);
  }
  {
    int n = D3_ * D_;
    build_wqkv_kernel<<<(n + 255) / 256, 256, 0, stream>>>(qw, kw, vw, wqkv);
  }

  gemm_bf16_kernel<0><<<dim3(D3_ / 128, BS_ / 128), 128, 0, stream>>>(
      x_bf, wqkv, nullptr, qkvb, BS_, D3_, D_);

  {
    int n = B_ * H_ * HD_ * S_;
    transpose_v_kernel<<<(n + 255) / 256, 256, 0, stream>>>(qkvb, vtb);
  }

  attn_kernel<<<dim3(S_ / 64, B_ * H_), 128, 0, stream>>>(qkvb, vtb, mask, wvb);

  gemm_bf16_kernel<0><<<dim3(D_ / 128, BS_ / 128), 128, 0, stream>>>(
      wvb, wo, nullptr, aob, BS_, D_, D_);

  gemm_bf16_kernel<1><<<dim3(F_ / 128, BS_ / 128), 128, 0, stream>>>(
      aob, wf1, f1b, h1b, BS_, F_, D_);

  gemm_bf16_kernel<2><<<dim3(D_ / 128, BS_ / 128), 128, 0, stream>>>(
      h1b, wf2, f2b, d_out, BS_, D_, F_);
}
